// PhotonicNeuralNetwork_40922448396368
// MI455X (gfx1250) — compile-verified, matrix operand order checked by reading, not run on hardware
//
#include <hip/hip_runtime.h>
#include <hip/hip_bf16.h>

// ---------------------------------------------------------------------------
// PhotonicNeuralNetwork on MI455X (gfx1250, wave32, WMMA)
// Layers: tanh(x) -> [tanh(x@W1^T+b1)+noise+thermal+coherence] -> same with W2
//         -> x@Wout^T + bout
// GEMMs: v_wmma_f32_16x16x32_f16, f32 accumulate, async global->LDS double
// buffering (one barrier per K-step), fused bias/tanh/noise/col-abs epilogue.
// ---------------------------------------------------------------------------

typedef _Float16 half_t;
typedef half_t v16h __attribute__((ext_vector_type(16)));
typedef half_t v8h  __attribute__((ext_vector_type(8)));
typedef float  v8f  __attribute__((ext_vector_type(8)));
typedef int    v4i_ __attribute__((ext_vector_type(4)));

#define NOISE_LEVEL 0.02f
#define CROSSTALK_C 0.05f
#define COHERENCE_C 0.03f

#define BM 128
#define BN 128
#define BK 32
#define LSTR 40                 // padded LDS row stride in halves (80B, 16B-aligned)
#define ATILE (BM * LSTR)
#define BTILE (BN * LSTR)

// ---- gfx1250 async global->LDS path (guarded; fallback = reg staging) -----
#if defined(__has_builtin)
#  if __has_builtin(__builtin_amdgcn_global_load_async_to_lds_b128) && \
      __has_builtin(__builtin_amdgcn_s_wait_asynccnt)
#    define USE_ASYNC_COPY 1
#  endif
#endif
#ifndef USE_ASYNC_COPY
#  define USE_ASYNC_COPY 0
#endif

#if USE_ASYNC_COPY
typedef __attribute__((address_space(1))) v4i_* as1_v4i_p;
typedef __attribute__((address_space(3))) v4i_* as3_v4i_p;
#endif

__device__ __forceinline__ void cp16B_to_lds(const half_t* g, half_t* l) {
#if USE_ASYNC_COPY
  __builtin_amdgcn_global_load_async_to_lds_b128(
      (as1_v4i_p)(void*)g, (as3_v4i_p)(void*)l, 0, 0);
#else
  *(uint4*)l = *(const uint4*)g;
#endif
}

__device__ __forceinline__ void lds_copy_fence() {
#if USE_ASYNC_COPY
  __builtin_amdgcn_s_wait_asynccnt(0);
#endif
}

// Deterministic hash -> approx uniform in [-1,1); stands in for JAX normals.
__device__ __forceinline__ float hash_noise(unsigned a, unsigned b, unsigned salt) {
  unsigned h = a * 0x9E3779B1u ^ (b + 0x7F4A7C15u) * 0x85EBCA77u ^ salt * 0xC2B2AE3Du;
  h ^= h >> 15; h *= 0x2C1B3C6Du; h ^= h >> 12; h *= 0x297A2D39u; h ^= h >> 15;
  return (float)(int)h * (1.0f / 2147483648.0f);
}

union FragH { v16h v; v8h h[2]; };

// ---------------------------------------------------------------------------
// Elementwise prep kernels
// ---------------------------------------------------------------------------
__global__ void k_tanh_to_h(const float* __restrict__ in, half_t* __restrict__ out, int n) {
  int i = blockIdx.x * blockDim.x + threadIdx.x;
  if (i < n) out[i] = (half_t)tanhf(in[i]);
}

__global__ void k_f32_to_h(const float* __restrict__ in, half_t* __restrict__ out, int n) {
  int i = blockIdx.x * blockDim.x + threadIdx.x;
  if (i < n) out[i] = (half_t)in[i];
}

__global__ void k_zero_f32(float* __restrict__ p, int n) {
  int i = blockIdx.x * blockDim.x + threadIdx.x;
  if (i < n) p[i] = 0.0f;
}

// ---------------------------------------------------------------------------
// Fused WMMA GEMM:  C[M,N] = tanh(A[M,K] @ B[N,K]^T + bias) + noise
// A,B f16 K-major. Block tile 128x128, 8 waves in 4(M)x2(N) grid, wave tile
// 32x64 (2x4 WMMA tiles). Double-buffered async LDS staging.
// Also atomically accumulates CROSSTALK * sum_rows |C[:,col]| into colabs.
// ---------------------------------------------------------------------------
__global__ void __launch_bounds__(256)
k_gemm_fused(const half_t* __restrict__ A, const half_t* __restrict__ B,
             const float* __restrict__ bias, half_t* __restrict__ C,
             float* __restrict__ colabs, int M, int N, int K, unsigned salt) {
  __shared__ half_t smem[2][ATILE + BTILE];

  const int t    = threadIdx.x;
  const int lane = t & 31;
  const int wid  = t >> 5;
  const int wm   = wid & 3;   // 4 M-waves  -> 32 rows each
  const int wn   = wid >> 2;  // 2 N-waves  -> 64 cols each
  const int m0   = blockIdx.y * BM;
  const int n0   = blockIdx.x * BN;

  // cooperative tile-load coordinates: both tiles are 128 rows x 32 halves,
  // each of the 256 threads moves 16 halves (2 x 16B) per tile, no divergence.
  const int lrowld = t >> 1;
  const int lcolld = (t & 1) * 16;
  const half_t* Ag = A + (size_t)(m0 + lrowld) * K + lcolld;
  const half_t* Bg = B + (size_t)(n0 + lrowld) * K + lcolld;

  // WMMA 16-bit A/B fragment addressing (wave32):
  // lane<16 -> K {0..7,16..23}; lane>=16 -> K {8..15,24..31}; row/col = lane&15
  const int lrow  = lane & 15;
  const int khalf = (lane >> 4) * 8;

  v8f acc[2][4] = {};

  // prologue: stage first K-slab into buffer 0
  {
    half_t* dA = &smem[0][lrowld * LSTR + lcolld];
    half_t* dB = &smem[0][ATILE + lrowld * LSTR + lcolld];
    cp16B_to_lds(Ag, dA);
    cp16B_to_lds(Ag + 8, dA + 8);
    cp16B_to_lds(Bg, dB);
    cp16B_to_lds(Bg + 8, dB + 8);
  }

  int buf = 0;
  for (int k0 = 0; k0 < K; k0 += BK) {
    lds_copy_fence();   // own async copies into smem[buf] complete
    __syncthreads();    // everyone's copies visible; prior reads of smem[buf^1] retired

    if (k0 + BK < K) {  // stage next K-slab into the other buffer (overlaps WMMAs)
      const int kn = k0 + BK;
      half_t* dA = &smem[buf ^ 1][lrowld * LSTR + lcolld];
      half_t* dB = &smem[buf ^ 1][ATILE + lrowld * LSTR + lcolld];
      cp16B_to_lds(Ag + kn, dA);
      cp16B_to_lds(Ag + kn + 8, dA + 8);
      cp16B_to_lds(Bg + kn, dB);
      cp16B_to_lds(Bg + kn + 8, dB + 8);
      if (kn + BK < K) __builtin_prefetch(Ag + kn + BK, 0, 1);
    }

    const half_t* sa = &smem[buf][0];
    const half_t* sb = &smem[buf][ATILE];

    FragH af[2], bfm[4];
#pragma unroll
    for (int mt = 0; mt < 2; ++mt) {
      const half_t* p = sa + (wm * 32 + mt * 16 + lrow) * LSTR + khalf;
      af[mt].h[0] = *(const v8h*)(p);
      af[mt].h[1] = *(const v8h*)(p + 16);
    }
#pragma unroll
    for (int nt = 0; nt < 4; ++nt) {
      const half_t* p = sb + (wn * 64 + nt * 16 + lrow) * LSTR + khalf;
      bfm[nt].h[0] = *(const v8h*)(p);
      bfm[nt].h[1] = *(const v8h*)(p + 16);
    }
#pragma unroll
    for (int mt = 0; mt < 2; ++mt)
#pragma unroll
      for (int nt = 0; nt < 4; ++nt)
        acc[mt][nt] = __builtin_amdgcn_wmma_f32_16x16x32_f16(
            false, af[mt].v, false, bfm[nt].v, (short)0, acc[mt][nt], false, false);

    buf ^= 1;
  }

  // Epilogue: bias + tanh + noise, store f16, per-column |x| reduction.
  // C/D layout: VGPR j -> row = base + j + 8*(lane>>4); col = base + (lane&15).
#pragma unroll
  for (int nt = 0; nt < 4; ++nt) {
    const int col = n0 + wn * 64 + nt * 16 + lrow;
    const float bv = bias[col];
    float asum = 0.0f;
#pragma unroll
    for (int mt = 0; mt < 2; ++mt) {
      const int rbase = m0 + wm * 32 + mt * 16 + ((lane >> 4) * 8);
#pragma unroll
      for (int j = 0; j < 8; ++j) {
        const int row = rbase + j;
        float v = tanhf(acc[mt][nt][j] + bv);
        v += hash_noise((unsigned)row, (unsigned)col, salt) * NOISE_LEVEL;
        C[(size_t)row * N + col] = (half_t)v;
        asum += fabsf(v);
      }
    }
    atomicAdd(&colabs[col], asum * CROSSTALK_C);  // layer_thermal contribution
  }
}

// ---------------------------------------------------------------------------
// Thermal EMA + 1/d^2 crosstalk matvec + pointwise apply
// ---------------------------------------------------------------------------
__global__ void k_thermal_update(const float* __restrict__ layer_thermal,
                                 const float* __restrict__ prev,
                                 float* __restrict__ out, int H) {
  int j = blockIdx.x * blockDim.x + threadIdx.x;
  if (j < H) out[j] = 0.7f * prev[j] + 0.3f * layer_thermal[j];
}

__global__ void k_crosstalk(const float* __restrict__ th, float* __restrict__ tn, int H) {
  int j = blockIdx.x * blockDim.x + threadIdx.x;
  if (j >= H) return;
  float s = 0.0f;
  for (int i = 0; i < H; ++i) {
    int d = i - j;
    if (d != 0) s += th[i] / (float)(d * d);
  }
  tn[j] = s * CROSSTALK_C;
}

__global__ void k_apply_thermal_coherence(half_t* __restrict__ h,
                                          const float* __restrict__ tn,
                                          int N, unsigned total, unsigned salt) {
  unsigned i = blockIdx.x * blockDim.x + threadIdx.x;
  if (i >= total) return;
  const int col = (int)(i % (unsigned)N);
  float v = (float)h[i] + tn[col];
  const float phase = hash_noise(i / (unsigned)N, (unsigned)col, salt) * COHERENCE_C;
  v = v + (v * __cosf(phase) - v) * COHERENCE_C;
  h[i] = (half_t)v;
}

// ---------------------------------------------------------------------------
// Final projection: out[M,2] = h[M,H] @ Wout[2,H]^T + bout   (wave per row)
// ---------------------------------------------------------------------------
__global__ void __launch_bounds__(256)
k_out_proj(const half_t* __restrict__ h, const float* __restrict__ Wout,
           const float* __restrict__ bout, float* __restrict__ out, int M, int H) {
  const int lane = threadIdx.x & 31;
  const int row  = blockIdx.x * (blockDim.x >> 5) + (threadIdx.x >> 5);
  if (row >= M) return;
  const half_t* hp = h + (size_t)row * H;
  float s0 = 0.0f, s1 = 0.0f;
  for (int k = lane; k < H; k += 32) {
    const float v = (float)hp[k];
    s0 += v * Wout[k];
    s1 += v * Wout[H + k];
  }
#pragma unroll
  for (int off = 16; off > 0; off >>= 1) {
    s0 += __shfl_down(s0, off, 32);
    s1 += __shfl_down(s1, off, 32);
  }
  if (lane == 0) {
    out[(size_t)row * 2 + 0] = s0 + bout[0];
    out[(size_t)row * 2 + 1] = s1 + bout[1];
  }
}

// ---------------------------------------------------------------------------
// Host launcher
// ---------------------------------------------------------------------------
extern "C" void kernel_launch(void* const* d_in, const int* in_sizes, int n_in,
                              void* d_out, int out_size, void* d_ws, size_t ws_size,
                              hipStream_t stream) {
  (void)in_sizes; (void)n_in; (void)out_size; (void)ws_size;
  const float* x    = (const float*)d_in[0];
  const float* W1   = (const float*)d_in[1];
  const float* b1   = (const float*)d_in[2];
  const float* W2   = (const float*)d_in[3];
  const float* b2   = (const float*)d_in[4];
  const float* Wout = (const float*)d_in[5];
  const float* bout = (const float*)d_in[6];
  float* out = (float*)d_out;

  const int Mb = 8192, K1 = 1024, H = 2048;

  char* w = (char*)d_ws;
  auto take = [&](size_t bytes) -> char* {
    char* p = w;
    w += (bytes + 255) & ~(size_t)255;
    return p;
  };
  half_t* xh   = (half_t*)take((size_t)Mb * K1 * 2);
  half_t* w1h  = (half_t*)take((size_t)H * K1 * 2);
  half_t* w2h  = (half_t*)take((size_t)H * H * 2);
  half_t* h1   = (half_t*)take((size_t)Mb * H * 2);
  half_t* h2   = (half_t*)take((size_t)Mb * H * 2);
  float* ca1 = (float*)take(H * 4);
  float* ca2 = (float*)take(H * 4);
  float* th0 = (float*)take(H * 4);
  float* th1 = (float*)take(H * 4);
  float* th2 = (float*)take(H * 4);
  float* tn1 = (float*)take(H * 4);
  float* tn2 = (float*)take(H * 4);

  const int TB = 256;
  // prep
  k_tanh_to_h<<<(Mb * K1 + TB - 1) / TB, TB, 0, stream>>>(x, xh, Mb * K1);
  k_f32_to_h<<<(H * K1 + TB - 1) / TB, TB, 0, stream>>>(W1, w1h, H * K1);
  k_f32_to_h<<<(H * H + TB - 1) / TB, TB, 0, stream>>>(W2, w2h, H * H);
  k_zero_f32<<<(3 * H + TB - 1) / TB, TB, 0, stream>>>(ca1, 3 * H);  // ca1, ca2, th0 contiguous

  const dim3 gemm_grid(H / BN, Mb / BM);

  // layer 1
  k_gemm_fused<<<gemm_grid, TB, 0, stream>>>(xh, w1h, b1, h1, ca1, Mb, H, K1, 0x1111u);
  k_thermal_update<<<H / TB, TB, 0, stream>>>(ca1, th0, th1, H);
  k_crosstalk<<<H / TB, TB, 0, stream>>>(th1, tn1, H);
  k_apply_thermal_coherence<<<(Mb * H) / TB, TB, 0, stream>>>(h1, tn1, H, (unsigned)(Mb * H), 0x2222u);

  // layer 2
  k_gemm_fused<<<gemm_grid, TB, 0, stream>>>(h1, w2h, b2, h2, ca2, Mb, H, H, 0x3333u);
  k_thermal_update<<<H / TB, TB, 0, stream>>>(ca2, th1, th2, H);
  k_crosstalk<<<H / TB, TB, 0, stream>>>(th2, tn2, H);
  k_apply_thermal_coherence<<<(Mb * H) / TB, TB, 0, stream>>>(h2, tn2, H, (unsigned)(Mb * H), 0x4444u);

  // output projection
  k_out_proj<<<Mb / (TB / 32), TB, 0, stream>>>(h2, Wout, bout, out, Mb, H);
}
